// LongformerSelfAttention_25391846654034
// MI455X (gfx1250) — compile-verified, matrix operand order checked by reading, not run on hardware
//
#include <hip/hip_runtime.h>
#include <hip/hip_fp16.h>

typedef __attribute__((ext_vector_type(16))) _Float16 v16h;
typedef __attribute__((ext_vector_type(8)))  float    v8f;
typedef __attribute__((ext_vector_type(4)))  int      v4i;

#define S_LEN  8192
#define HID    768
#define NHEAD  12
#define HDIM   64
#define WINW   256

union F16Frag { v16h v; uint4 q[2]; };
union H8Pack  { _Float16 h[8]; uint4 u; };
union H4Pack  { _Float16 h[4]; uint2 u; };

static __device__ __forceinline__ v8f wmma_f16(const F16Frag& a, const F16Frag& b, v8f c) {
  // D = A(16x32 f16) * B(32x16 f16) + C(16x16 f32)
  return __builtin_amdgcn_wmma_f32_16x16x32_f16(false, a.v, false, b.v, (short)0, c, false, false);
}

// ---- async global->LDS (gfx1250), guarded so absence falls back cleanly ----
#if defined(__has_builtin)
#if __has_builtin(__builtin_amdgcn_global_load_async_to_lds_b128)
#define HAVE_ASYNC_LDS 1
#endif
#endif

#ifdef HAVE_ASYNC_LDS
typedef __attribute__((address_space(1))) v4i gv4i;   // global int4*
typedef __attribute__((address_space(3))) v4i lv4i;   // LDS int4*
static __device__ __forceinline__ void cp16_g2l(const _Float16* g, _Float16* l) {
  // flat->AS1 keeps the 64-bit address; flat->AS3 keeps the low-32 LDS offset
  __builtin_amdgcn_global_load_async_to_lds_b128(
      (gv4i*)(size_t)g, (lv4i*)(unsigned)(size_t)l, 0, 0);
}
static __device__ __forceinline__ void async_wait0() {
#if __has_builtin(__builtin_amdgcn_s_wait_asynccnt)
  __builtin_amdgcn_s_wait_asynccnt(0);
#else
  asm volatile("s_wait_asynccnt 0x0" ::: "memory");
#endif
}
#else
static __device__ __forceinline__ void cp16_g2l(const _Float16* g, _Float16* l) {
  *(uint4*)l = *(const uint4*)g;   // VGPR round-trip fallback
}
static __device__ __forceinline__ void async_wait0() {}
#endif

// ---------------------------------------------------------------------------
// Kernel 1: X (f32) -> Xh (f16), vectorized 4-wide
// ---------------------------------------------------------------------------
__global__ void convert_x_kernel(const float* __restrict__ X, _Float16* __restrict__ Xh) {
  const int i = blockIdx.x * blockDim.x + threadIdx.x;  // exactly S*HID/4 threads
  const float4 f = ((const float4*)X)[i];
  H4Pack p;
  p.h[0] = (_Float16)f.x; p.h[1] = (_Float16)f.y;
  p.h[2] = (_Float16)f.z; p.h[3] = (_Float16)f.w;
  ((uint2*)Xh)[i] = p.u;
}

// ---------------------------------------------------------------------------
// Kernel 2: W[k][n] (f32) -> Wt[n][k] (f16), LDS-tiled transpose, 3 matrices
// ---------------------------------------------------------------------------
__global__ void wtrans_kernel(const float* __restrict__ Wq, const float* __restrict__ Wk,
                              const float* __restrict__ Wv, _Float16* __restrict__ Wt) {
  __shared__ _Float16 tile[32][33];
  const float* src = blockIdx.z == 0 ? Wq : (blockIdx.z == 1 ? Wk : Wv);
  _Float16* dst = Wt + (size_t)blockIdx.z * HID * HID;
  const int n0 = blockIdx.x * 32;
  const int k0 = blockIdx.y * 32;
  const int tx = threadIdx.x, ty = threadIdx.y;
  #pragma unroll
  for (int i = 0; i < 4; ++i)
    tile[ty + 8 * i][tx] = (_Float16)src[(size_t)(k0 + ty + 8 * i) * HID + n0 + tx];
  __syncthreads();
  #pragma unroll
  for (int i = 0; i < 4; ++i)
    dst[(size_t)(n0 + ty + 8 * i) * HID + k0 + tx] = tile[tx][ty + 8 * i];
}

// ---------------------------------------------------------------------------
// Kernel 3: QKV projection GEMM (unchanged; codegen verified good)
// ---------------------------------------------------------------------------
__global__ __launch_bounds__(256) void qkv_gemm_kernel(
    const _Float16* __restrict__ Xh, const _Float16* __restrict__ Wt,
    const float* __restrict__ bq, const float* __restrict__ bk, const float* __restrict__ bv,
    _Float16* __restrict__ Qh, _Float16* __restrict__ Kh, _Float16* __restrict__ Vt)
{
  const int proj = blockIdx.z;
  const _Float16* Wp = Wt + (size_t)proj * HID * HID;
  const int lane = threadIdx.x & 31;
  const int w    = threadIdx.x >> 5;
  const int ln16 = lane & 15;
  const int hl   = lane >> 4;
  const int kb8  = hl * 8;
  const int Mb = blockIdx.x * 64 + (w >> 2) * 32;
  const int Nb = blockIdx.y * 128 + (w & 3) * 32;

  v8f zero = {};
  v8f acc[2][2] = {{zero, zero}, {zero, zero}};

  const _Float16* arow[2] = { Xh + (size_t)(Mb + ln16) * HID,
                              Xh + (size_t)(Mb + 16 + ln16) * HID };
  const _Float16* brow[2] = { Wp + (size_t)(Nb + ln16) * HID,
                              Wp + (size_t)(Nb + 16 + ln16) * HID };

  for (int kk = 0; kk < HID; kk += 32) {
    F16Frag a[2], b[2];
    #pragma unroll
    for (int i = 0; i < 2; ++i) {
      a[i].q[0] = *(const uint4*)(arow[i] + kk + kb8);
      a[i].q[1] = *(const uint4*)(arow[i] + kk + kb8 + 16);
      b[i].q[0] = *(const uint4*)(brow[i] + kk + kb8);
      b[i].q[1] = *(const uint4*)(brow[i] + kk + kb8 + 16);
      __builtin_prefetch(arow[i] + kk + 64, 0, 3);   // -> global_prefetch_b8
      __builtin_prefetch(brow[i] + kk + 64, 0, 3);
    }
    #pragma unroll
    for (int mi = 0; mi < 2; ++mi)
      #pragma unroll
      for (int ni = 0; ni < 2; ++ni)
        acc[mi][ni] = wmma_f16(a[mi], b[ni], acc[mi][ni]);
  }

  const float* bias = proj == 0 ? bq : (proj == 1 ? bk : bv);
  const float qscale = proj == 0 ? 0.125f : 1.0f;   // 1/sqrt(64) folded into Q
  #pragma unroll
  for (int mi = 0; mi < 2; ++mi) {
    #pragma unroll
    for (int ni = 0; ni < 2; ++ni) {
      const int n = Nb + ni * 16 + ln16;
      const float bn = bias[n];
      if (proj == 2) {
        H8Pack p;
        #pragma unroll
        for (int r = 0; r < 8; ++r) p.h[r] = (_Float16)(acc[mi][ni][r] + bn);
        const int srow = Mb + mi * 16 + 8 * hl;
        *(uint4*)(Vt + (size_t)n * S_LEN + srow) = p.u;
      } else {
        _Float16* dst = proj == 0 ? Qh : Kh;
        #pragma unroll
        for (int r = 0; r < 8; ++r) {
          const int row = Mb + mi * 16 + r + 8 * hl;
          dst[(size_t)row * HID + n] = (_Float16)((acc[mi][ni][r] + bn) * qscale);
        }
      }
    }
  }
}

// ---------------------------------------------------------------------------
// Kernel 4: banded attention, flash-style, with double-buffered ASYNC K tiles.
//   grid (64, 12), block 256 (8 waves). WG owns 128 query rows; key-band union
//   = 640 keys = 5 superblocks of 128, staged global->LDS asynchronously.
// ---------------------------------------------------------------------------
#define KROWS  128   // keys per superblock
#define KPAD   72    // padded LDS row stride (halves): 144 B, conflict-free b128
#define NSUPER 5

__global__ __launch_bounds__(256) void attn_kernel(
    const _Float16* __restrict__ Qh, const _Float16* __restrict__ Kh,
    const _Float16* __restrict__ Vt, float* __restrict__ out)
{
  __shared__ __align__(16) _Float16 kbuf[2][KROWS * KPAD];  // 36 KB
  __shared__ __align__(16) _Float16 pscr[8][16 * 48];       // 12 KB

  const int qb0  = blockIdx.x * 128;        // first query row of this WG
  const int h    = blockIdx.y;
  const int tid  = threadIdx.x;
  const int lane = tid & 31;
  const int wid  = tid >> 5;                // 0..7
  const int ln16 = lane & 15;
  const int hl   = lane >> 4;
  const int kb8  = hl * 8;
  const int qb   = qb0 + wid * 16;          // first query row of this wave

  // ---- cooperative K superblock staging: 128 rows x 64 halves -> LDS ----
  auto stage_k = [&](int t, _Float16* buf) {
    const int kgbase = qb0 - WINW + KROWS * t;
    #pragma unroll
    for (int j = 0; j < 4; ++j) {
      const int idx = tid + j * 256;        // 1024 x 16 B = one 16 KB tile
      const int row = idx >> 3;             // 0..127
      const int ch  = idx & 7;              // 16-B chunk within row
      int key = kgbase + row;
      key = key < 0 ? 0 : (key >= S_LEN ? S_LEN - 1 : key);   // clamped rows get masked
      cp16_g2l(Kh + (size_t)key * HID + h * HDIM + ch * 8, buf + row * KPAD + ch * 8);
    }
  };

  // Q A-fragments (d = 0..31 and 32..63)
  F16Frag aq0, aq1;
  {
    const _Float16* qp = Qh + (size_t)(qb + ln16) * HID + h * HDIM;
    aq0.q[0] = *(const uint4*)(qp + kb8);
    aq0.q[1] = *(const uint4*)(qp + kb8 + 16);
    aq1.q[0] = *(const uint4*)(qp + 32 + kb8);
    aq1.q[1] = *(const uint4*)(qp + 32 + kb8 + 16);
  }

  v8f zero = {};
  v8f oacc[4] = {zero, zero, zero, zero};
  float m[8], l[8];
  #pragma unroll
  for (int r = 0; r < 8; ++r) { m[r] = -3.0e38f; l[r] = 0.f; }

  stage_k(0, kbuf[0]);
  async_wait0();
  __syncthreads();

  const int ylo = wid * 16;                 // band (WG-local keys) = [ylo, ylo+527]
  const int ybstart = ylo & ~31;
  const int ybend   = ylo + 528;

  for (int t = 0; t < NSUPER; ++t) {
    const _Float16* kb = kbuf[t & 1];
    if (t + 1 < NSUPER) stage_k(t + 1, kbuf[(t + 1) & 1]);   // overlap with compute

    const int lo = (t * KROWS > ybstart) ? t * KROWS : ybstart;
    const int hi = ((t + 1) * KROWS < ybend) ? (t + 1) * KROWS : ybend;
    for (int yb = lo; yb < hi; yb += 32) {
      const int kg = qb0 - WINW + yb;       // global key base of this 32-block
      if (kg + 32 <= 0 || kg >= S_LEN) continue;
      const int rb = yb - t * KROWS;        // row base inside the staged buffer

      // ---- scores from LDS K fragments ----
      v8f s0 = zero, s1 = zero;
      {
        const _Float16* kp = kb + (size_t)(rb + ln16) * KPAD + kb8;
        F16Frag blo, bhi;
        blo.q[0] = *(const uint4*)(kp);
        blo.q[1] = *(const uint4*)(kp + 16);
        bhi.q[0] = *(const uint4*)(kp + 32);
        bhi.q[1] = *(const uint4*)(kp + 48);
        s0 = wmma_f16(aq0, blo, s0);
        s0 = wmma_f16(aq1, bhi, s0);
      }
      {
        const _Float16* kp = kb + (size_t)(rb + 16 + ln16) * KPAD + kb8;
        F16Frag blo, bhi;
        blo.q[0] = *(const uint4*)(kp);
        blo.q[1] = *(const uint4*)(kp + 16);
        bhi.q[0] = *(const uint4*)(kp + 32);
        bhi.q[1] = *(const uint4*)(kp + 48);
        s1 = wmma_f16(aq0, blo, s1);
        s1 = wmma_f16(aq1, bhi, s1);
      }

      // ---- mask + online softmax ----
      const int kp0 = kg + ln16;
      const int kp1 = kp0 + 16;
      float e0[8], e1[8], sc[8];
      #pragma unroll
      for (int r = 0; r < 8; ++r) {
        const int gq = qb + r + 8 * hl;
        const bool v0 = (kp0 >= 0) && (kp0 < S_LEN) && (kp0 >= gq - WINW) && (kp0 <= gq + WINW);
        const bool v1 = (kp1 >= 0) && (kp1 < S_LEN) && (kp1 >= gq - WINW) && (kp1 <= gq + WINW);
        const float x0 = v0 ? s0[r] : -3.0e38f;
        const float x1 = v1 ? s1[r] : -3.0e38f;
        float rmax = fmaxf(x0, x1);
        #pragma unroll
        for (int d = 1; d < 16; d <<= 1) rmax = fmaxf(rmax, __shfl_xor(rmax, d, 32));
        const float mn = fmaxf(m[r], rmax);
        if (mn <= -1.0e38f) { sc[r] = 1.f; e0[r] = 0.f; e1[r] = 0.f; }
        else {
          sc[r] = __expf(m[r] - mn);
          e0[r] = __expf(x0 - mn);
          e1[r] = __expf(x1 - mn);
        }
        m[r] = mn;
        float psum = e0[r] + e1[r];
        #pragma unroll
        for (int d = 1; d < 16; d <<= 1) psum += __shfl_xor(psum, d, 32);
        l[r] = l[r] * sc[r] + psum;
      }
      #pragma unroll
      for (int tt = 0; tt < 4; ++tt) {
        #pragma unroll
        for (int r = 0; r < 8; ++r) oacc[tt][r] *= sc[r];
      }

      // ---- P: C-layout f32 -> A-layout f16 via per-wave LDS scratch ----
      _Float16* ps = &pscr[wid][0];
      #pragma unroll
      for (int r = 0; r < 8; ++r) {
        const int M = r + 8 * hl;
        ps[M * 48 + ln16]      = (_Float16)e0[r];
        ps[M * 48 + 16 + ln16] = (_Float16)e1[r];
      }
      F16Frag ap;  // same-wave LDS ops are in-order; no barrier needed
      {
        const _Float16* pr = ps + ln16 * 48 + kb8;
        ap.q[0] = *(const uint4*)(pr);
        ap.q[1] = *(const uint4*)(pr + 16);
      }

      // ---- O += P(16x32) @ V(32x64): Vt[n][s] gives contiguous B-frags ----
      const int k0 = kg + kb8;
      const int k0c = k0 < 0 ? 0 : (k0 > S_LEN - 8 ? S_LEN - 8 : k0);
      const int k1 = k0 + 16;
      const int k1c = k1 < 0 ? 0 : (k1 > S_LEN - 8 ? S_LEN - 8 : k1);
      #pragma unroll
      for (int tt = 0; tt < 4; ++tt) {
        const _Float16* vp = Vt + (size_t)(h * HDIM + tt * 16 + ln16) * S_LEN;
        F16Frag bv;
        bv.q[0] = *(const uint4*)(vp + k0c);
        bv.q[1] = *(const uint4*)(vp + k1c);
        oacc[tt] = wmma_f16(ap, bv, oacc[tt]);
      }
    }

    async_wait0();      // next superblock staged
    __syncthreads();    // everyone done reading current buffer
  }

  // ---- epilogue: out = O / l, f32 ----
  #pragma unroll
  for (int r = 0; r < 8; ++r) {
    const float il = l[r] > 0.f ? 1.f / l[r] : 0.f;
    const int row = qb + r + 8 * hl;
    float* op = out + (size_t)row * HID + h * HDIM + ln16;
    #pragma unroll
    for (int t = 0; t < 4; ++t) op[t * 16] = oacc[t][r] * il;
  }
}

// ---------------------------------------------------------------------------
extern "C" void kernel_launch(void* const* d_in, const int* in_sizes, int n_in,
                              void* d_out, int out_size, void* d_ws, size_t ws_size,
                              hipStream_t stream) {
  (void)in_sizes; (void)n_in; (void)out_size; (void)ws_size;
  const float* X  = (const float*)d_in[0];
  const float* Wq = (const float*)d_in[1];
  const float* bq = (const float*)d_in[2];
  const float* Wk = (const float*)d_in[3];
  const float* bk = (const float*)d_in[4];
  const float* Wv = (const float*)d_in[5];
  const float* bv = (const float*)d_in[6];
  float* out = (float*)d_out;

  char* ws = (char*)d_ws;
  _Float16* Xh = (_Float16*)ws; ws += (size_t)S_LEN * HID * sizeof(_Float16);
  _Float16* Wt = (_Float16*)ws; ws += (size_t)3 * HID * HID * sizeof(_Float16);
  _Float16* Qh = (_Float16*)ws; ws += (size_t)S_LEN * HID * sizeof(_Float16);
  _Float16* Kh = (_Float16*)ws; ws += (size_t)S_LEN * HID * sizeof(_Float16);
  _Float16* Vt = (_Float16*)ws; ws += (size_t)S_LEN * HID * sizeof(_Float16);

  convert_x_kernel<<<(S_LEN * HID / 4) / 256, 256, 0, stream>>>(X, Xh);
  wtrans_kernel<<<dim3(HID / 32, HID / 32, 3), dim3(32, 8), 0, stream>>>(Wq, Wk, Wv, Wt);
  qkv_gemm_kernel<<<dim3(S_LEN / 64, HID / 128, 3), 256, 0, stream>>>(
      Xh, Wt, bq, bk, bv, Qh, Kh, Vt);
  attn_kernel<<<dim3(S_LEN / 128, NHEAD), 256, 0, stream>>>(Qh, Kh, Vt, out);
}
